// GNN_16793322128022
// MI455X (gfx1250) — matrix operand order repaired and checked by reading, not compiled, not run on hardware
//
#include <hip/hip_runtime.h>
#include <hip/hip_bf16.h>

typedef __attribute__((ext_vector_type(16))) __bf16        v16bf;
typedef __attribute__((ext_vector_type(2)))  __bf16        v2bf;
typedef __attribute__((ext_vector_type(8)))  float         v8f;
typedef __attribute__((ext_vector_type(16))) unsigned short v16u;
typedef __attribute__((ext_vector_type(8)))  unsigned int  v8u;
typedef __attribute__((ext_vector_type(4)))  unsigned int  u32x4;
typedef __attribute__((ext_vector_type(8)))  int           i32x8;
typedef __attribute__((ext_vector_type(4)))  int           i32x4;
typedef int v4i __attribute__((vector_size(16)));           // matches builtin proto

// ---- CDNA5 feature probes (compile-safe: fall back if builtin absent) ----
#if defined(__AMDGCN__) && __has_builtin(__builtin_amdgcn_global_load_async_to_lds_b128) && \
    __has_builtin(__builtin_amdgcn_s_wait_asynccnt)
#define GNN_ASYNC 1
#else
#define GNN_ASYNC 0
#endif

#if defined(__AMDGCN__) && __has_builtin(__builtin_amdgcn_tensor_load_to_lds) && \
    __has_builtin(__builtin_amdgcn_s_wait_tensorcnt)
#define GNN_TDM 1
#if __has_include(<hip/amd_detail/amd_gfx1250_TDM.h>)
#define GNN_TDM6 1   // therock headers -> 6-arg builtin
#else
#define GNN_TDM6 0   // ROCm 7.2 -> 5-arg builtin
#endif
#else
#define GNN_TDM 0
#endif

__device__ __forceinline__ unsigned short f2bf(float f) {
    unsigned int u = __builtin_bit_cast(unsigned int, f);
    u += 0x7FFFu + ((u >> 16) & 1u);          // round-to-nearest-even
    return (unsigned short)(u >> 16);
}

__device__ __forceinline__ v16bf cvt16(const float t[16]) {
#if defined(__AMDGCN__) && __has_builtin(__builtin_amdgcn_cvt_pk_bf16_f32)
    union { v16bf v; v2bf p[8]; } u;
#pragma unroll
    for (int i = 0; i < 8; ++i)
        u.p[i] = __builtin_amdgcn_cvt_pk_bf16_f32(t[2 * i], t[2 * i + 1]);
    return u.v;
#else
    v16u ua;
#pragma unroll
    for (int i = 0; i < 16; ++i) ua[i] = f2bf(t[i]);
    return __builtin_bit_cast(v16bf, ua);
#endif
}

// ---------------------------------------------------------------------------
// Pack a [128 x 128] row-major f32 weight matrix (K x Ncol) into bf16 WMMA
// B-fragments. Fragment (kt, ct): K in [kt*32, kt*32+32), col in [ct*16,+16).
// Lane L holds 16 halves: col = ct*16 + (L&15); element e -> K = kt*32 +
// 16*(L>>4) + e.  Stored at out[((kt*8+ct)*32 + L)*16 + e]  (32B/lane).
// ---------------------------------------------------------------------------
__global__ void pack_w128(const float* __restrict__ W,
                          unsigned short* __restrict__ out) {
    int idx  = blockIdx.x * 256 + threadIdx.x;   // 0..16383
    int e    = idx & 15;
    int lane = (idx >> 4) & 31;
    int fr   = idx >> 9;                         // fragment id 0..31
    int kt = fr >> 3, ct = fr & 7;
    int col = ct * 16 + (lane & 15);
    int k   = kt * 32 + ((lane >> 4) << 4) + e;
    out[idx] = f2bf(W[k * 128 + col]);
}

// ---------------------------------------------------------------------------
// C = act(bn(A @ B + bias)) [+ resid],  A:[M,128] f32 (optionally
// (1+eps)*A1 + A2), B packed bf16 fragments.  8 waves/block, 128 rows/block,
// each wave: 16x128 strip, 32x v_wmma_f32_16x16x32_bf16.
// B fragments (32KB) are DMA'd once per block into LDS by the Tensor Data
// Mover (wave 0, TENSORcnt), then read via ds_load_b128.
// ---------------------------------------------------------------------------
__global__ void __launch_bounds__(256)
gemm128_bf16(const float* __restrict__ A1, const float* __restrict__ A2,
             const float* __restrict__ epsPtr,
             const unsigned short* __restrict__ Bp,
             const float* __restrict__ bias, const float* __restrict__ gamma,
             const float* __restrict__ beta, const float* __restrict__ resid,
             float* __restrict__ Out, int M, int doRelu) {
    const int wave = threadIdx.x >> 5;
    const int lane = threadIdx.x & 31;
    const int half = lane >> 4;
    const int R    = blockIdx.x * 128 + wave * 16;
    const int rlane = R + (lane & 15);
    const int rc    = (rlane < M) ? rlane : (M - 1);
    const float epsv = epsPtr ? (1.0f + *epsPtr) : 1.0f;

#if GNN_TDM
    __shared__ __align__(128) unsigned short sB[16384];   // 32 KB
    if (wave == 0) {
        unsigned long long ga = (unsigned long long)Bp;
        unsigned ldsa = (unsigned)(unsigned long long)
            (__attribute__((address_space(3))) unsigned short*)sB;
        u32x4 g0 = {0u, 0u, 0u, 0u};
        g0[0] = 1u;                                   // count=1, user mode
        g0[1] = ldsa;                                 // lds_addr
        g0[2] = (unsigned)ga;                         // global_addr[31:0]
        g0[3] = (unsigned)(ga >> 32) | (2u << 30);    // global_addr[56:32] | type=2
        i32x8 g1 = {0, 0, 0, 0, 0, 0, 0, 0};
        g1[0] = (int)(3u << 16);        // workgroup_mask=0, data_size=3 (8B)
        g1[1] = (int)(4096u << 16);     // tensor_dim0[15:0] = 4096 (8B units)
        g1[2] = (int)(1u << 16);        // tensor_dim0 hi = 0, tensor_dim1 = 1
        g1[3] = (int)(4096u << 16);     // tile_dim0 = 4096  (1-D, 32 KB)
        g1[5] = 4096;                   // tensor_dim0_stride
        i32x4 z4 = {0, 0, 0, 0};
#if GNN_TDM6
        i32x8 z8 = {0, 0, 0, 0, 0, 0, 0, 0};
        __builtin_amdgcn_tensor_load_to_lds(g0, g1, z4, z4, z8, 0);
#else
        __builtin_amdgcn_tensor_load_to_lds(g0, g1, z4, z4, 0);
#endif
        __builtin_amdgcn_s_wait_tensorcnt(0);
    }
    __syncthreads();
    const unsigned short* Bsrc = sB;
#else
    const unsigned short* Bsrc = Bp;
#endif

    // ---- A fragments (contiguous layout: e -> K=16*half+e, matching packed B)
    v16bf afrag[4];
#pragma unroll
    for (int kt = 0; kt < 4; ++kt) {
        const float* p = A1 + (size_t)rc * 128 + kt * 32 + half * 16;
        float4 f0 = *(const float4*)(p);
        float4 f1 = *(const float4*)(p + 4);
        float4 g0 = *(const float4*)(p + 8);
        float4 g1 = *(const float4*)(p + 12);
        float t[16] = {f0.x, f0.y, f0.z, f0.w, f1.x, f1.y, f1.z, f1.w,
                       g0.x, g0.y, g0.z, g0.w, g1.x, g1.y, g1.z, g1.w};
        if (A2) {
            const float* q = A2 + (size_t)rc * 128 + kt * 32 + half * 16;
            float4 a0 = *(const float4*)(q);
            float4 a1 = *(const float4*)(q + 4);
            float4 b0 = *(const float4*)(q + 8);
            float4 b1 = *(const float4*)(q + 12);
            float u[16] = {a0.x, a0.y, a0.z, a0.w, a1.x, a1.y, a1.z, a1.w,
                           b0.x, b0.y, b0.z, b0.w, b1.x, b1.y, b1.z, b1.w};
#pragma unroll
            for (int i = 0; i < 16; ++i) t[i] = epsv * t[i] + u[i];
        }
        afrag[kt] = cvt16(t);
    }

    v8f c[8] = {};
#pragma unroll
    for (int kt = 0; kt < 4; ++kt) {
#pragma unroll
        for (int ct = 0; ct < 8; ++ct) {
            v16bf bfrag = __builtin_bit_cast(
                v16bf, *(const v8u*)(Bsrc + ((size_t)((kt * 8 + ct) * 32 + lane)) * 16));
            c[ct] = __builtin_amdgcn_wmma_f32_16x16x32_bf16(
                false, afrag[kt], false, bfrag, (short)0, c[ct], false, false);
        }
    }

    // ---- epilogue: y = acc*scale + shift ; relu ; + resid
#pragma unroll
    for (int ct = 0; ct < 8; ++ct) {
        int col = ct * 16 + (lane & 15);
        float scale = 1.0f, shift = 0.0f;
        if (gamma) scale = gamma[col] * rsqrtf(1.0f + 1e-5f);
        if (beta)  shift = beta[col];
        if (bias)  shift += bias[col] * scale;
#pragma unroll
        for (int v = 0; v < 8; ++v) {
            int row = R + half * 8 + v;
            if (row < M) {
                float val = c[ct][v] * scale + shift;
                if (doRelu) val = fmaxf(val, 0.0f);
                if (resid)  val += resid[(size_t)row * 128 + col];
                Out[(size_t)row * 128 + col] = val;
            }
        }
    }
}

// ---------------------------------------------------------------------------
// Fused edge encoder + GINE message + scatter:
//   ea  = relu(bn(edge_attr @ eW + eb))           (K=16 GEMV, VALU)
//   m   = relu(h[src] + ea);  agg[dst] += m       (atomic, L2-resident)
// 64 edges per block; edge_attr staged into LDS with async copy (ASYNCcnt);
// 128 threads cover one edge's feature row -> coalesced 512B gathers/atomics.
// ---------------------------------------------------------------------------
#define EPB 64
__global__ void __launch_bounds__(256)
edge_msg(const float* __restrict__ h, const int* __restrict__ ei,
         const float* __restrict__ eattr,
         const float* __restrict__ eW, const float* __restrict__ eb,
         const float* __restrict__ eg, const float* __restrict__ ebeta,
         float* __restrict__ agg, int E) {
    __shared__ __align__(16) float sattr[EPB * 16];
    const int t    = threadIdx.x;
    const int f    = t & 127;
    const int esub = t >> 7;
    const int e0   = blockIdx.x * EPB;
    const int rem  = E - e0;

    if (rem >= EPB) {
#if GNN_ASYNC
        __attribute__((address_space(1))) v4i* g =
            (__attribute__((address_space(1))) v4i*)(eattr + (size_t)e0 * 16) + t;
        __attribute__((address_space(3))) v4i* l =
            (__attribute__((address_space(3))) v4i*)sattr + t;
        __builtin_amdgcn_global_load_async_to_lds_b128(g, l, 0, 0);
        __builtin_amdgcn_s_wait_asynccnt(0);
#else
        ((float4*)sattr)[t] = ((const float4*)(eattr + (size_t)e0 * 16))[t];
#endif
    } else {
        for (int i = t; i < EPB * 16; i += 256)
            sattr[i] = (i < rem * 16) ? eattr[(size_t)e0 * 16 + i] : 0.0f;
    }

    float wcol[16];
#pragma unroll
    for (int k = 0; k < 16; ++k) wcol[k] = eW[k * 128 + f];
    const float scale = eg[f] * rsqrtf(1.0f + 1e-5f);
    const float shift = eb[f] * scale + ebeta[f];
    __syncthreads();

    for (int j = esub; j < EPB; j += 2) {
        int e = e0 + j;
        if (e >= E) break;
        int src = ei[e];
        int dst = ei[E + e];
        float acc = 0.0f;
#pragma unroll
        for (int k = 0; k < 16; ++k) acc += sattr[j * 16 + k] * wcol[k];
        float ea = fmaxf(acc * scale + shift, 0.0f);
        float m  = fmaxf(h[(size_t)src * 128 + f] + ea, 0.0f);
        atomicAdd(&agg[(size_t)dst * 128 + f], m);
    }
}

// ---------------------------------------------------------------------------
// pooled[batch[n]] += h[n]   (512x128 output, L2-resident atomics)
// ---------------------------------------------------------------------------
__global__ void pool_sum(const float* __restrict__ h, const int* __restrict__ batch,
                         float* __restrict__ pooled, int Nn) {
    int idx = blockIdx.x * 256 + threadIdx.x;
    int n = idx >> 7, f = idx & 127;
    if (n < Nn)
        atomicAdd(&pooled[(size_t)batch[n] * 128 + f], h[(size_t)n * 128 + f]);
}

extern "C" void kernel_launch(void* const* d_in, const int* in_sizes, int n_in,
                              void* d_out, int out_size, void* d_ws, size_t ws_size,
                              hipStream_t stream) {
    const float* x       = (const float*)d_in[0];
    const int*   ei      = (const int*)d_in[1];
    const float* eattr   = (const float*)d_in[2];
    const int*   batch   = (const int*)d_in[3];
    const float* in_W    = (const float*)d_in[4];
    const float* in_b    = (const float*)d_in[5];
    const float* in_g    = (const float*)d_in[6];
    const float* in_beta = (const float*)d_in[7];
    const float* edge_W  = (const float*)d_in[8];
    const float* edge_b  = (const float*)d_in[9];
    const float* edge_g  = (const float*)d_in[10];
    const float* edge_be = (const float*)d_in[11];
    const float* conv_W  = (const float*)d_in[12];
    const float* eps     = (const float*)d_in[13];
    const float* bn_g    = (const float*)d_in[14];
    const float* bn_b    = (const float*)d_in[15];
    const float* out_W1  = (const float*)d_in[16];
    const float* out_b1  = (const float*)d_in[17];
    const float* out_g1  = (const float*)d_in[18];
    const float* out_be1 = (const float*)d_in[19];
    const float* out_W2  = (const float*)d_in[20];
    const float* out_b2  = (const float*)d_in[21];

    const int Nn = in_sizes[0] / 128;   // 50000 nodes
    const int E  = in_sizes[2] / 16;    // 1.6M edges

    // ---- workspace carve ----
    char* ws = (char*)d_ws;
    unsigned short* pw = (unsigned short*)ws;        // 7 * 16384 bf16 halves
    size_t pwBytes = ((7 * 16384 * sizeof(unsigned short)) + 255) & ~(size_t)255;
    float* hA     = (float*)(ws + pwBytes);
    float* hB     = hA + (size_t)Nn * 128;
    float* agg    = hB + (size_t)Nn * 128;
    float* pooled = agg + (size_t)Nn * 128;
    float* tbuf   = pooled + 512 * 128;

    unsigned short* pw_in   = pw;
    unsigned short* pw_conv = pw + 16384;            // 4 matrices
    unsigned short* pw_o1   = pw + 5 * 16384;
    unsigned short* pw_o2   = pw + 6 * 16384;

    dim3 b256(256);

    // ---- pack weights to bf16 WMMA-B fragment layout ----
    pack_w128<<<64, b256, 0, stream>>>(in_W, pw_in);
    for (int l = 0; l < 4; ++l)
        pack_w128<<<64, b256, 0, stream>>>(conv_W + (size_t)l * 16384,
                                           pw_conv + (size_t)l * 16384);
    pack_w128<<<64, b256, 0, stream>>>(out_W1, pw_o1);
    pack_w128<<<64, b256, 0, stream>>>(out_W2, pw_o2);

    const int gblocks = (Nn + 127) / 128;
    const int eblocks = (E + EPB - 1) / EPB;

    // ---- input encoder: h = relu(bn(x @ in_W + in_b)) ----
    gemm128_bf16<<<gblocks, b256, 0, stream>>>(x, nullptr, nullptr, pw_in,
        in_b, in_g, in_beta, nullptr, hA, Nn, 1);

    float* hin = hA;
    float* hout = hB;
    for (int l = 0; l < 4; ++l) {
        (void)hipMemsetAsync(agg, 0, (size_t)Nn * 128 * sizeof(float), stream);
        edge_msg<<<eblocks, b256, 0, stream>>>(hin, ei, eattr,
            edge_W + (size_t)l * 16 * 128, edge_b + l * 128,
            edge_g + l * 128, edge_be + l * 128, agg, E);
        // h = relu(bn(((1+eps)h + agg) @ conv_W)) + h
        gemm128_bf16<<<gblocks, b256, 0, stream>>>(hin, agg, eps + l,
            pw_conv + (size_t)l * 16384, nullptr, bn_g + l * 128, bn_b + l * 128,
            hin, hout, Nn, 1);
        float* tmp = hin; hin = hout; hout = tmp;
    }

    // ---- pooling ----
    (void)hipMemsetAsync(pooled, 0, 512 * 128 * sizeof(float), stream);
    pool_sum<<<(Nn * 128 + 255) / 256, b256, 0, stream>>>(hin, batch, pooled, Nn);

    // ---- output MLP ----
    gemm128_bf16<<<4, b256, 0, stream>>>(pooled, nullptr, nullptr, pw_o1,
        out_b1, out_g1, out_be1, nullptr, tbuf, 512, 1);
    gemm128_bf16<<<4, b256, 0, stream>>>(tbuf, nullptr, nullptr, pw_o2,
        out_b2, nullptr, nullptr, nullptr, (float*)d_out, 512, 0);
}
